// MultiHeadAttention_85246510891612
// MI455X (gfx1250) — matrix operand order repaired and checked by reading, not compiled, not run on hardware
//
#include <hip/hip_runtime.h>

// ---------------------------------------------------------------------------
// Types for CDNA5 WMMA (wave32): bf16 A/B fragments, f32 accumulator.
// ---------------------------------------------------------------------------
typedef __attribute__((ext_vector_type(16))) __bf16 bf16x16;
typedef __attribute__((ext_vector_type(8)))  __bf16 bf16x8;
typedef __attribute__((ext_vector_type(8)))  float  f32x8;

#define D_MODEL 1024
#define N_HEAD  16
#define D_HEAD  64
#define BATCH   4
#define SEQ     2048
#define NTOK    (BATCH * SEQ)       // 8192
#define QKV_N   (3 * D_MODEL)       // 3072

// ---------------------------------------------------------------------------
// CDNA5 primitives
// ---------------------------------------------------------------------------
__device__ __forceinline__ f32x8 wmma_bf16(bf16x16 a, bf16x16 b, f32x8 c) {
    return __builtin_amdgcn_wmma_f32_16x16x32_bf16(
        /*neg_a=*/false, a, /*neg_b=*/false, b,
        /*c_mod=*/(short)0, c, /*reuse_a=*/false, /*reuse_b=*/false);
}

__device__ __forceinline__ bf16x16 join8(bf16x8 lo, bf16x8 hi) {
    return __builtin_shufflevector(lo, hi, 0,1,2,3,4,5,6,7,8,9,10,11,12,13,14,15);
}

typedef __attribute__((address_space(3))) void lds_void;
__device__ __forceinline__ unsigned lds_off(void* p) {
    return (unsigned)(size_t)(lds_void*)p;   // generic -> LDS addrspace offset
}

// Async 16-byte copy global -> LDS (ASYNCcnt-tracked, no VGPR data movement).
__device__ __forceinline__ void async_copy16(const __bf16* g, __bf16* l) {
    asm volatile("global_load_async_to_lds_b128 %0, %1, off"
                 :: "v"(lds_off(l)), "v"((unsigned long long)(size_t)g)
                 : "memory");
}

__device__ __forceinline__ void wait_async_and_barrier() {
    asm volatile("s_wait_asynccnt 0x0" ::: "memory");
    __syncthreads();
}

// A fragment: 16x32 bf16 tile, row-major source with leading dim `ld`.
// Lane layout (ISA 7.12.2): m = lane&15, half = lane>>4;
// element e: K = (e>>3)*16 + half*8 + (e&7)  -> two contiguous 8-elt runs.
__device__ __forceinline__ bf16x16 load_a_frag(const __bf16* base, int ld, int lane) {
    int m    = lane & 15;
    int half = lane >> 4;
    const __bf16* p = base + (size_t)m * ld + half * 8;
    bf16x8 lo = *(const bf16x8*)(p);
    bf16x8 hi = *(const bf16x8*)(p + 16);
    return join8(lo, hi);
}

// B fragment: 32x16 bf16 tile (KxN). Source stored N-major (B^T row-major,
// leading dim `ld`): lane n reads K runs [half*8,+8) and [16+half*8,+8).
__device__ __forceinline__ bf16x16 load_b_frag(const __bf16* base, int ld, int lane) {
    int n    = lane & 15;
    int half = lane >> 4;
    const __bf16* p = base + (size_t)n * ld + half * 8;
    bf16x8 lo = *(const bf16x8*)(p);
    bf16x8 hi = *(const bf16x8*)(p + 16);
    return join8(lo, hi);
}

// Wave32 row reductions across the 16-lane halves (C-layout rows span 16 lanes)
__device__ __forceinline__ float rmax16(float v) {
    v = fmaxf(v, __shfl_xor(v, 1, 32));
    v = fmaxf(v, __shfl_xor(v, 2, 32));
    v = fmaxf(v, __shfl_xor(v, 4, 32));
    v = fmaxf(v, __shfl_xor(v, 8, 32));
    return v;
}
__device__ __forceinline__ float rsum16(float v) {
    v += __shfl_xor(v, 1, 32);
    v += __shfl_xor(v, 2, 32);
    v += __shfl_xor(v, 4, 32);
    v += __shfl_xor(v, 8, 32);
    return v;
}

// ---------------------------------------------------------------------------
// Conversion kernels
// ---------------------------------------------------------------------------
__global__ __launch_bounds__(256) void cvt_f32_to_bf16(const float* __restrict__ src,
                                                       __bf16* __restrict__ dst, int n) {
    int i = blockIdx.x * 256 + threadIdx.x;
    if (i < n) dst[i] = (__bf16)src[i];
}

// src: [R][C] f32 row-major -> dst: [C][R] bf16 (transpose)
__global__ __launch_bounds__(256) void cvt_transpose_bf16(const float* __restrict__ src,
                                                          __bf16* __restrict__ dst,
                                                          int R, int C) {
    int i = blockIdx.x * 256 + threadIdx.x;
    if (i < R * C) {
        int r = i / C, c = i % C;
        dst[(size_t)c * R + r] = (__bf16)src[i];
    }
}

// ---------------------------------------------------------------------------
// Double-buffered async-LDS GEMM mainloop.
// Block = 128 threads (4 waves) computes a 128x64 output tile.
// Per K=32 step: async-stage a 128x32 A tile (8 KB) + 64x32 B tile (4 KB)
// into LDS while WMMAs consume the other stage. Each wave owns 32 M-rows x
// 64 N (2 A frags, 8 accumulators, 8 WMMAs per step; B shared by all waves).
// LDS: 2 stages x 12 KB = 24 KB of the 320 KB/WGP.
// ---------------------------------------------------------------------------
#define A_STAGE 4096   // elements per A stage (128*32)
#define B_STAGE 2048   // elements per B stage (64*32)

__device__ __forceinline__ void stage_fill(const __bf16* __restrict__ Ag,
                                           const __bf16* __restrict__ Bg,
                                           __bf16* As, __bf16* Bs,
                                           int lda, int ldb, int tid) {
    // A tile: 128 rows x 32 cols; each thread copies one row (4 x 16B)
#pragma unroll
    for (int it = 0; it < 4; ++it)
        async_copy16(Ag + (size_t)tid * lda + it * 8, As + tid * 32 + it * 8);
    // B tile: 64 rows x 32 cols; two threads per row (2 x 16B each)
    const int row = tid >> 1;
    const int c0  = (tid & 1) * 8;
#pragma unroll
    for (int it = 0; it < 2; ++it)
        async_copy16(Bg + (size_t)row * ldb + c0 + it * 16,
                     Bs + row * 32 + c0 + it * 16);
}

__device__ __forceinline__ void gemm_mainloop(const __bf16* __restrict__ Ag,  // @ (mblock, 0)
                                              const __bf16* __restrict__ Bg,  // @ (nblock, 0)
                                              int Kdim, int lda, int ldb,
                                              __bf16* As, __bf16* Bs,
                                              f32x8 acc[2][4]) {
    const int tid  = threadIdx.x;
    const int lane = tid & 31;
    const int wave = tid >> 5;

    stage_fill(Ag, Bg, As, Bs, lda, ldb, tid);
    wait_async_and_barrier();

    int s = 0;
    for (int k = 0; k < Kdim; k += 32) {
        const int ns = s ^ 1;
        if (k + 32 < Kdim)   // issue next stage; overlaps with WMMAs below
            stage_fill(Ag + (k + 32), Bg + (k + 32),
                       As + ns * A_STAGE, Bs + ns * B_STAGE, lda, ldb, tid);

        const __bf16* as = As + s * A_STAGE + wave * 32 * 32;
        bf16x16 a0 = load_a_frag(as,           32, lane);
        bf16x16 a1 = load_a_frag(as + 16 * 32, 32, lane);
#pragma unroll
        for (int j = 0; j < 4; ++j) {
            bf16x16 b = load_b_frag(Bs + s * B_STAGE + j * 16 * 32, 32, lane);
            acc[0][j] = wmma_bf16(a0, b, acc[0][j]);
            acc[1][j] = wmma_bf16(a1, b, acc[1][j]);
        }
        wait_async_and_barrier();
        s = ns;
    }
}

// ---------------------------------------------------------------------------
// QKV GEMM: qkv[8192,3072] = Xbf @ Wqkv^T + b, scattered into
//   Q [b,h,t,d] (pre-scaled by 1/sqrt(64)), K [b,h,t,d], Vt [b,h,d,t], bf16.
// ---------------------------------------------------------------------------
__global__ __launch_bounds__(128) void gemm_qkv(const __bf16* __restrict__ X,
                                                const __bf16* __restrict__ Wt,   // [3072][1024]
                                                const float*  __restrict__ bias, // [3072]
                                                __bf16* __restrict__ Q,
                                                __bf16* __restrict__ K,
                                                __bf16* __restrict__ Vt) {
    __shared__ __align__(16) __bf16 As[2 * A_STAGE];
    __shared__ __align__(16) __bf16 Bs[2 * B_STAGE];

    const int lane  = threadIdx.x & 31;
    const int wave  = threadIdx.x >> 5;
    const int mbase = blockIdx.y * 128 + wave * 32;
    const int nbase = blockIdx.x * 64;
    const int half  = lane >> 4;
    const int nl    = lane & 15;

    f32x8 acc[2][4] = {{f32x8{}, f32x8{}, f32x8{}, f32x8{}},
                       {f32x8{}, f32x8{}, f32x8{}, f32x8{}}};
    gemm_mainloop(X + (size_t)(blockIdx.y * 128) * D_MODEL,
                  Wt + (size_t)nbase * D_MODEL,
                  D_MODEL, D_MODEL, D_MODEL, As, Bs, acc);

#pragma unroll
    for (int sub = 0; sub < 2; ++sub)
#pragma unroll
    for (int j = 0; j < 4; ++j) {
        const int ncol  = nbase + j * 16 + nl;
        const int which = ncol >> 10;       // 0=Q 1=K 2=V
        const int rem   = ncol & 1023;
        const int h     = rem >> 6;
        const int d     = rem & 63;
        const float bv  = bias[ncol];
#pragma unroll
        for (int e = 0; e < 8; ++e) {
            const int r = mbase + sub * 16 + e + 8 * half;   // global token
            const int b = r >> 11;
            const int t = r & 2047;
            const float v = acc[sub][j][e] + bv;
            const size_t bh = (size_t)(b * N_HEAD + h);
            if (which == 0)      Q [(bh * SEQ + t) * D_HEAD + d] = (__bf16)(v * 0.125f);
            else if (which == 1) K [(bh * SEQ + t) * D_HEAD + d] = (__bf16)v;
            else                 Vt[(bh * D_HEAD + d) * SEQ + t] = (__bf16)v;
        }
    }
}

// ---------------------------------------------------------------------------
// Flash attention: grid = B*H*(SEQ/64) blocks of 128 threads; each wave owns a
// 16-query tile. Key blocks of 32; next block's K fragments prefetched before
// the softmax VALU work so global latency hides behind __expf/shfl.
// ---------------------------------------------------------------------------
__global__ __launch_bounds__(128) void flash_attn(const __bf16* __restrict__ Q,
                                                  const __bf16* __restrict__ K,
                                                  const __bf16* __restrict__ Vt,
                                                  __bf16* __restrict__ ctx) { // [8192][1024]
    __shared__ __align__(16) __bf16 plds[4][16 * 32];   // per-wave 16x32 P tile

    const int lane = threadIdx.x & 31;
    const int wave = threadIdx.x >> 5;
    const int half = lane >> 4;
    const int nl   = lane & 15;

    const int gid    = blockIdx.x;
    const int qsuper = gid & 31;          // SEQ/64 = 32
    const int bh     = gid >> 5;          // b*16 + h
    const int qbase  = qsuper * 64 + wave * 16;

    const size_t hoff = (size_t)bh * SEQ * D_HEAD;
    const __bf16* Qh = Q  + hoff;
    const __bf16* Kh = K  + hoff;
    const __bf16* Vh = Vt + hoff;         // [64][2048]

    const bf16x16 aq0 = load_a_frag(Qh + (size_t)qbase * D_HEAD + 0,  D_HEAD, lane);
    const bf16x16 aq1 = load_a_frag(Qh + (size_t)qbase * D_HEAD + 32, D_HEAD, lane);

    f32x8 o[4] = {f32x8{}, f32x8{}, f32x8{}, f32x8{}};
    float mrow[8], lrow[8];
#pragma unroll
    for (int e = 0; e < 8; ++e) { mrow[e] = -1e30f; lrow[e] = 0.0f; }

    __bf16* pw = &plds[wave][0];

    // prime K fragments for kb = 0
    bf16x16 kf0 = load_b_frag(Kh + 0,            D_HEAD, lane);
    bf16x16 kf1 = load_b_frag(Kh + 32,           D_HEAD, lane);
    bf16x16 kf2 = load_b_frag(Kh + 16 * 64,      D_HEAD, lane);
    bf16x16 kf3 = load_b_frag(Kh + 16 * 64 + 32, D_HEAD, lane);

    for (int kb = 0; kb < SEQ; kb += 32) {
        // ---- scores: two 16x16 tiles over keys [kb,kb+16) and [kb+16,kb+32)
        f32x8 s0 = {};
        s0 = wmma_bf16(aq0, kf0, s0);
        s0 = wmma_bf16(aq1, kf1, s0);
        f32x8 s1 = {};
        s1 = wmma_bf16(aq0, kf2, s1);
        s1 = wmma_bf16(aq1, kf3, s1);

        // ---- prefetch next key block's K fragments (hide behind softmax)
        const int kn = (kb + 32 < SEQ) ? kb + 32 : 0;
        const __bf16* kn0 = Kh + (size_t)kn * D_HEAD;
        const __bf16* kn1 = Kh + (size_t)(kn + 16) * D_HEAD;
        bf16x16 nf0 = load_b_frag(kn0 + 0,  D_HEAD, lane);
        bf16x16 nf1 = load_b_frag(kn0 + 32, D_HEAD, lane);
        bf16x16 nf2 = load_b_frag(kn1 + 0,  D_HEAD, lane);
        bf16x16 nf3 = load_b_frag(kn1 + 32, D_HEAD, lane);

        // ---- issue V fragment loads for current block early
        bf16x16 vf[4];
#pragma unroll
        for (int j = 0; j < 4; ++j)
            vf[j] = load_b_frag(Vh + (size_t)(j * 16) * SEQ + kb, SEQ, lane);

        // ---- online softmax (rows live across 16-lane halves)
        float corr[8];
#pragma unroll
        for (int e = 0; e < 8; ++e) {
            float mx = rmax16(fmaxf(s0[e], s1[e]));
            float nm = fmaxf(mrow[e], mx);
            corr[e]  = __expf(mrow[e] - nm);
            mrow[e]  = nm;
            float p0 = __expf(s0[e] - nm);
            float p1 = __expf(s1[e] - nm);
            lrow[e]  = lrow[e] * corr[e] + rsum16(p0 + p1);
            s0[e] = p0;
            s1[e] = p1;
        }
#pragma unroll
        for (int j = 0; j < 4; ++j)
#pragma unroll
            for (int e = 0; e < 8; ++e) o[j][e] *= corr[e];

        // ---- P (C-layout f32) -> LDS bf16 (row-major 16x32) -> A fragment
#pragma unroll
        for (int e = 0; e < 8; ++e) {
            const int r = e + 8 * half;
            pw[r * 32 + nl]      = (__bf16)s0[e];
            pw[r * 32 + 16 + nl] = (__bf16)s1[e];
        }
        const bf16x16 pf = load_a_frag(pw, 32, lane);

        // ---- O += P @ V  (V stored [d][t]: contiguous per-lane B loads)
#pragma unroll
        for (int j = 0; j < 4; ++j) o[j] = wmma_bf16(pf, vf[j], o[j]);

        kf0 = nf0; kf1 = nf1; kf2 = nf2; kf3 = nf3;
    }

    // ---- epilogue: normalize and emit context in [b, t, h*64+d] bf16
    const int b = bh >> 4;
    const int h = bh & 15;
    float inv[8];
#pragma unroll
    for (int e = 0; e < 8; ++e) inv[e] = 1.0f / lrow[e];
#pragma unroll
    for (int j = 0; j < 4; ++j)
#pragma unroll
        for (int e = 0; e < 8; ++e) {
            const int t = qbase + e + 8 * half;
            ctx[((size_t)(b * SEQ + t)) * D_MODEL + h * D_HEAD + j * 16 + nl] =
                (__bf16)(o[j][e] * inv[e]);
        }
}

// ---------------------------------------------------------------------------
// Output GEMM: out[8192,1024] = ctx @ Wout^T + b_out (f32 result)
// ---------------------------------------------------------------------------
__global__ __launch_bounds__(128) void gemm_out(const __bf16* __restrict__ A,    // ctx bf16
                                                const __bf16* __restrict__ Wt,   // [1024][1024]
                                                const float*  __restrict__ bias,
                                                float* __restrict__ out) {
    __shared__ __align__(16) __bf16 As[2 * A_STAGE];
    __shared__ __align__(16) __bf16 Bs[2 * B_STAGE];

    const int lane  = threadIdx.x & 31;
    const int wave  = threadIdx.x >> 5;
    const int mbase = blockIdx.y * 128 + wave * 32;
    const int nbase = blockIdx.x * 64;
    const int half  = lane >> 4;
    const int nl    = lane & 15;

    f32x8 acc[2][4] = {{f32x8{}, f32x8{}, f32x8{}, f32x8{}},
                       {f32x8{}, f32x8{}, f32x8{}, f32x8{}}};
    gemm_mainloop(A + (size_t)(blockIdx.y * 128) * D_MODEL,
                  Wt + (size_t)nbase * D_MODEL,
                  D_MODEL, D_MODEL, D_MODEL, As, Bs, acc);

#pragma unroll
    for (int sub = 0; sub < 2; ++sub)
#pragma unroll
    for (int j = 0; j < 4; ++j) {
        const int ncol = nbase + j * 16 + nl;
        const float bv = bias[ncol];
#pragma unroll
        for (int e = 0; e < 8; ++e) {
            const int r = mbase + sub * 16 + e + 8 * half;
            out[(size_t)r * D_MODEL + ncol] = acc[sub][j][e] + bv;
        }
    }
}

// ---------------------------------------------------------------------------
// Host-side launch
// ---------------------------------------------------------------------------
extern "C" void kernel_launch(void* const* d_in, const int* in_sizes, int n_in,
                              void* d_out, int out_size, void* d_ws, size_t ws_size,
                              hipStream_t stream) {
    const float* x     = (const float*)d_in[0];
    const float* w_qkv = (const float*)d_in[1];
    const float* b_qkv = (const float*)d_in[2];
    const float* w_out = (const float*)d_in[3];
    const float* b_out = (const float*)d_in[4];
    float* out = (float*)d_out;

    char* ws = (char*)d_ws;
    size_t off = 0;
    __bf16* x_bf   = (__bf16*)(ws + off); off += (size_t)NTOK * D_MODEL * 2;   // 16 MB
    __bf16* wqkv_t = (__bf16*)(ws + off); off += (size_t)QKV_N * D_MODEL * 2;  //  6 MB
    __bf16* wout_t = (__bf16*)(ws + off); off += (size_t)D_MODEL * D_MODEL * 2;//  2 MB
    __bf16* Qb     = (__bf16*)(ws + off); off += (size_t)NTOK * D_MODEL * 2;   // 16 MB
    __bf16* Kb     = (__bf16*)(ws + off); off += (size_t)NTOK * D_MODEL * 2;   // 16 MB
    __bf16* Vtb    = (__bf16*)(ws + off); off += (size_t)NTOK * D_MODEL * 2;   // 16 MB
    __bf16* ctx    = (__bf16*)(ws + off); off += (size_t)NTOK * D_MODEL * 2;   // 16 MB

    // 1) precision conversion / weight transposition
    {
        int n = NTOK * D_MODEL;
        cvt_f32_to_bf16<<<(n + 255) / 256, 256, 0, stream>>>(x, x_bf, n);
    }
    {
        int n = D_MODEL * QKV_N;
        cvt_transpose_bf16<<<(n + 255) / 256, 256, 0, stream>>>(w_qkv, wqkv_t,
                                                                D_MODEL, QKV_N);
    }
    {
        int n = D_MODEL * D_MODEL;
        cvt_transpose_bf16<<<(n + 255) / 256, 256, 0, stream>>>(w_out, wout_t,
                                                                D_MODEL, D_MODEL);
    }

    // 2) fused QKV projection -> Q (pre-scaled), K, V^T
    {
        dim3 grid(QKV_N / 64, NTOK / 128);
        gemm_qkv<<<grid, 128, 0, stream>>>(x_bf, wqkv_t, b_qkv, Qb, Kb, Vtb);
    }

    // 3) flash attention -> ctx [b, t, h*d]
    {
        int blocks = BATCH * N_HEAD * (SEQ / 64);
        flash_attn<<<blocks, 128, 0, stream>>>(Qb, Kb, Vtb, ctx);
    }

    // 4) output projection (f32 out + bias)
    {
        dim3 grid(D_MODEL / 64, NTOK / 128);
        gemm_out<<<grid, 128, 0, stream>>>(ctx, wout_t, b_out, out);
    }
}
